// CrossDFN_25812753449764
// MI455X (gfx1250) — compile-verified, matrix operand order checked by reading, not run on hardware
//
#include <hip/hip_runtime.h>

// ---------------- CDNA5 types ----------------
typedef __attribute__((ext_vector_type(16))) __bf16 v16bf;
typedef __attribute__((ext_vector_type(8)))  float  v8f;
typedef __attribute__((ext_vector_type(4)))  unsigned int v4u;
typedef __attribute__((ext_vector_type(8)))  int v8i;
typedef __attribute__((ext_vector_type(4)))  int v4i;

union BF16x16 { v16bf v; uint4 q[2]; };
static_assert(sizeof(v16bf) == 32, "v16bf must be 32 bytes");

#if defined(__has_builtin)
#if __has_builtin(__builtin_amdgcn_tensor_load_to_lds) && __has_builtin(__builtin_amdgcn_s_wait_tensorcnt)
#define HAVE_TDM 1
#endif
#endif
#ifndef HAVE_TDM
#define HAVE_TDM 0
#endif

// ---------------- problem constants ----------------
#define NN 2
#define CC 64
#define HH 64
#define WW 64
#define WP 80                  // padded width for unfold gathers
#define PAD 8                  // >= max dilated reach (5)
#define ELEMS (NN*HH*WW*CC)    // 524288

__constant__ int dKS[5] = {5,5,3,3,3};
__constant__ int dRS[5] = {1,2,3,4,5};
// per-scale B-fragment element base: layout [s][tap][side][nb][f][lane][j]
__constant__ int dSBASE[5] = {0, 204800, 409600, 483328, 557056};

// ---------------- workspace layout (bytes) ----------------
static const size_t ACT_OFF   = 0;          // 10 * 524288 bf16 = 10,485,760
static const size_t BW_OFF    = 10485760;   // 630784 bf16      =  1,261,568
static const size_t COMBX_OFF = 11747328;   // 524288 f32
static const size_t COMBY_OFF = 13844480;   // 524288 f32
static const size_t XTP_OFF   = 15941632;   // 2*64*80*64 f32   =  2,621,440
static const size_t YTP_OFF   = 18563072;
static const size_t FW_OFF    = 21184512;   // 73728 bf16       =    147,456
// memset region: combx..ytp end = 9,437,184 bytes

// ---------------- helpers ----------------
__device__ inline v8f wmma_bf16(v16bf a, v16bf b, v8f c) {
  return __builtin_amdgcn_wmma_f32_16x16x32_bf16(
      false, a, false, b, (short)0, c, false, false);
}

// A-matrix 16x32 bf16 per ISA lane layout (lane&15 = M row, lane>>4 = K sub-chunk)
__device__ inline v16bf load_a_frag(const __bf16* __restrict__ row, int hi, int f) {
  BF16x16 u;
  u.q[0] = *(const uint4*)(row + f*32 + 8*hi);
  u.q[1] = *(const uint4*)(row + f*32 + 16 + 8*hi);
  return u.v;
}

// B fragments pre-swizzled: 32 contiguous bytes per lane
__device__ inline v16bf load_b_frag(const __bf16* p) {
  BF16x16 u;
  const uint4* q = (const uint4*)p;
  u.q[0] = q[0];
  u.q[1] = q[1];
  return u.v;
}

// TDM: DMA one contiguous 16 KB block global -> LDS (D# per ISA §8.3/8.4)
__device__ inline void tdm_stage_16k(const void* gsrc, void* lds_dst) {
#if HAVE_TDM
  unsigned long long ga = (unsigned long long)gsrc;
  unsigned lds_off = (unsigned)(unsigned long long)lds_dst;  // flat low 32 = LDS offset
  // group0: count=1 | lds_addr | global_addr[56:0] | type=2
  v4u g0 = { 1u, lds_off, (unsigned)ga,
             (unsigned)((ga >> 32) & 0x01FFFFFFull) | (2u << 30) };
  // group1: data_size=4B; tensor_dim0=tile_dim0=4096 dwords; dim1=1; stride0=4096
  const unsigned n4 = 4096u;
  v8i g1 = { (int)(2u << 16),                 // data_size=2 (4 bytes)
             (int)((n4 & 0xFFFFu) << 16),     // tensor_dim0[15:0]
             (int)(((n4 >> 16) & 0xFFFFu) | (1u << 16)), // tensor_dim0[31:16], tensor_dim1=1
             (int)((n4 & 0xFFFFu) << 16),     // tile_dim0
             1,                               // tile_dim1 = 1
             (int)n4,                         // tensor_dim0_stride
             0, 0 };
  v4i z4 = {0,0,0,0};
  v8i z8 = {0,0,0,0,0,0,0,0};
  // 6-arg form (this toolchain): (g0, g1, g2, g3, g4, cpol)
  __builtin_amdgcn_tensor_load_to_lds(g0, g1, z4, z4, z8, 0);
  __builtin_amdgcn_s_wait_tensorcnt(0);
#else
  (void)gsrc; (void)lds_dst;
#endif
}

// cooperative 16 KB stage: all 256 threads copy 64 B each (fallback / no-TDM)
__device__ inline void coop_stage_16k(const void* gsrc, uint4* lds_dst) {
  const uint4* src = (const uint4*)gsrc;
  const int i = threadIdx.x * 4;
#pragma unroll
  for (int q = 0; q < 4; ++q) lds_dst[i + q] = src[i + q];
}

// ---------------- prep kernels ----------------
__global__ __launch_bounds__(256) void k_transpose_pad(
    const float* __restrict__ in, float* __restrict__ out) {
  int t = blockIdx.x*256 + threadIdx.x;
  if (t >= ELEMS) return;
  int c = t & 63, w = (t>>6)&63, h = (t>>12)&63, n = t>>18;
  out[((n*HH + h)*WP + PAD + w)*CC + c] = in[((n*CC + c)*HH + h)*WW + w];
}

// depthwise conv (dilated, same pad) + BN + ReLU -> bf16 NHWC act
__global__ __launch_bounds__(256) void k_prep_act(
    const float* __restrict__ inT, const float* __restrict__ dw,
    const float* __restrict__ bns, const float* __restrict__ bnb,
    __bf16* __restrict__ act, int k, int r) {
  int t = blockIdx.x*256 + threadIdx.x;
  if (t >= ELEMS) return;
  int c = t & 63, w = (t>>6)&63, h = (t>>12)&63, n = t>>18;
  int p = r*(k-1)/2;
  float s = 0.f;
  for (int i = 0; i < k; ++i) {
    int ih = h + i*r - p;
    if ((unsigned)ih >= (unsigned)HH) continue;
    const float* row = inT + ((size_t)(n*HH + ih)*WP + PAD + w)*CC + c;
    for (int j = 0; j < k; ++j)
      s += row[(size_t)(j*r - p)*CC] * dw[(c*k + i)*k + j];   // horizontal pad is zero
  }
  s = fmaxf(s*bns[c] + bnb[c], 0.f);
  act[t] = (__bf16)s;
}

// pointwise weight [C*k*k, C] -> B frags; out base already includes side offset;
// per-tap stride 8192 elems (both sides interleaved -> one contiguous 16 KB per tap)
__global__ __launch_bounds__(256) void k_prep_bw(
    const float* __restrict__ pww, __bf16* __restrict__ out, int k) {
  int idx = blockIdx.x*256 + threadIdx.x;
  int kk2 = k*k;
  if (idx >= kk2*4096) return;
  int j    = idx & 15;
  int lane = (idx >> 4) & 31;
  int f    = (idx >> 9) & 1;
  int nb   = (idx >> 10) & 3;
  int tap  = idx >> 12;
  int col = lane & 15, hi = lane >> 4;
  int co = nb*16 + col;                                         // N col (c_out)
  int ci = f*32 + ((j < 8) ? (8*hi + j) : (16 + 8*hi + (j-8))); // K row (c_in)
  out[(size_t)tap*8192 + (idx & 4095)] = (__bf16)pww[(co*kk2 + tap)*CC + ci];
}

// fuse_w [64,128,3,3] -> B frags, layout ((tap*4 + kc)*4 + nb)*512 + lane*16 + j
__global__ __launch_bounds__(256) void k_prep_fw(
    const float* __restrict__ fw, __bf16* __restrict__ out) {
  int idx = blockIdx.x*256 + threadIdx.x;
  if (idx >= 73728) return;
  int j    = idx & 15;
  int lane = (idx >> 4) & 31;
  int nb   = (idx >> 9) & 3;
  int kc   = (idx >> 11) & 3;
  int tap  = idx >> 13;
  int col = lane & 15, hi = lane >> 4;
  int oc = nb*16 + col;
  int ic = kc*32 + ((j < 8) ? (8*hi + j) : (16 + 8*hi + (j-8)));
  int ky = tap/3, kx = tap%3;
  out[idx] = (__bf16)fw[((oc*128 + ic)*3 + ky)*3 + kx];
}

// ---------------- fused dynamic-filter kernel ----------------
struct DfnArgs {
  const __bf16* act;     // 10 x [N,H,W,C] bf16
  const __bf16* bw;      // B fragments [s][tap][side][nb][f][lane][j]
  const float*  xt;      // x NHWC f32, padded width WP
  const float*  yt;      // y NHWC f32, padded width WP
  float*        combx;   // [N,H,W,C] f32 accumulators (zeroed)
  float*        comby;
  const float*  pwb[10]; // pointwise biases (C*k*k,)
};

// grid: blockIdx.x = s*64 + tb; 8 waves/block share scale s, wave handles one 16-pixel tile
__global__ __launch_bounds__(256) void k_dfn_fused(DfnArgs a) {
  __shared__ uint4 sB4[1024];                // 16 KB per-tap B fragments (both sides)
  const int s    = blockIdx.x >> 6;          // 0..4
  const int tb   = blockIdx.x & 63;
  const int wave = threadIdx.x >> 5;
  const int lane = threadIdx.x & 31;
  const int tile = tb*8 + wave;              // 0..511
  const int k = dKS[s], r = dRS[s], p = r*(k-1)/2, kk2 = k*k;
  const int w0 = (tile & 3) << 4;
  const int h  = (tile >> 2) & 63;
  const int n  = tile >> 8;
  const int col = lane & 15, hi = lane >> 4;
  const int gy = 2*s, gx = 2*s + 1;

  const __bf16* actY = a.act + (size_t)gy*ELEMS + ((size_t)((n*HH + h)*WW) + (w0 + col))*CC;
  const __bf16* actX = a.act + (size_t)gx*ELEMS + ((size_t)((n*HH + h)*WW) + (w0 + col))*CC;
  v16bf ay0 = load_a_frag(actY, hi, 0), ay1 = load_a_frag(actY, hi, 1);
  v16bf ax0 = load_a_frag(actX, hi, 0), ax1 = load_a_frag(actX, hi, 1);

  const __bf16* bwS = a.bw + dSBASE[s];
  const __bf16* sb  = (const __bf16*)sB4;
  const float* pbY = a.pwb[gy];
  const float* pbX = a.pwb[gx];

  const v8f zf = {0.f,0.f,0.f,0.f,0.f,0.f,0.f,0.f};
  float cx[4][8] = {};
  float cy[4][8] = {};

  for (int t = 0; t < kk2; ++t) {
    // ---- stage tap t's 16 KB of B fragments into LDS ----
#if HAVE_TDM
    if (threadIdx.x == 0) {
      if (t + 1 < kk2)
        __builtin_prefetch((const void*)(bwS + (size_t)(t+1)*8192), 0, 1);
      tdm_stage_16k((const void*)(bwS + (size_t)t*8192), (void*)sB4);
    }
#else
    coop_stage_16k((const void*)(bwS + (size_t)t*8192), sB4);
#endif
    __syncthreads();

    const int dy = (t / k) * r - p;
    const int dx = (t % k) * r - p;
    const int hh = h + dy;
    if ((unsigned)hh < (unsigned)HH) {       // wave-uniform (OOB rows contribute zero)
      // padded rows: all per-lane gathers unconditional
      const float* xrow = a.xt + ((size_t)(n*HH + hh)*WP + PAD + w0 + dx)*CC;
      const float* yrow = a.yt + ((size_t)(n*HH + hh)*WP + PAD + w0 + dx)*CC;
#pragma unroll
      for (int nb = 0; nb < 4; ++nb) {
        const int c = nb*16 + col;
        const __bf16* sy = sb +        nb*1024 + lane*16;
        const __bf16* sx = sb + 4096 + nb*1024 + lane*16;
        {  // y-branch kernel * unfold(x)
          v8f acc = zf;
          acc = wmma_bf16(ay0, load_b_frag(sy),       acc);
          acc = wmma_bf16(ay1, load_b_frag(sy + 512), acc);
          const float bias = pbY[c*kk2 + t];
#pragma unroll
          for (int v = 0; v < 8; ++v) {
            const int m = v + 8*hi;
            cx[nb][v] += (acc[v] + bias) * xrow[(size_t)m*CC + c];
          }
        }
        {  // x-branch kernel * unfold(y)
          v8f acc = zf;
          acc = wmma_bf16(ax0, load_b_frag(sx),       acc);
          acc = wmma_bf16(ax1, load_b_frag(sx + 512), acc);
          const float bias = pbX[c*kk2 + t];
#pragma unroll
          for (int v = 0; v < 8; ++v) {
            const int m = v + 8*hi;
            cy[nb][v] += (acc[v] + bias) * yrow[(size_t)m*CC + c];
          }
        }
      }
    }
    __syncthreads();
  }

  // accumulate this scale's contribution into combined_{x,y}
#pragma unroll
  for (int nb = 0; nb < 4; ++nb) {
    const int c = nb*16 + col;
#pragma unroll
    for (int v = 0; v < 8; ++v) {
      const int m = v + 8*hi;
      const size_t o = ((size_t)((n*HH + h)*WW) + (w0 + m))*CC + c;
      unsafeAtomicAdd(a.combx + o, cx[nb][v]);
      unsafeAtomicAdd(a.comby + o, cy[nb][v]);
    }
  }
}

// ---------------- fuse 3x3 conv (128->64) via WMMA + BN/ReLU ----------------
__global__ __launch_bounds__(256) void k_fuse_conv(
    const float* __restrict__ combx, const float* __restrict__ comby,
    const __bf16* __restrict__ fw,
    const float* __restrict__ nxs, const float* __restrict__ nxb,
    const float* __restrict__ nys, const float* __restrict__ nyb,
    const float* __restrict__ bns, const float* __restrict__ bnb,
    float* __restrict__ out) {
  __shared__ uint4 sB4[1024];                 // 16 KB per-tap fuse B fragments
  const int wave = threadIdx.x >> 5;
  const int lane = threadIdx.x & 31;
  const int wid  = blockIdx.x*8 + wave;       // 0..511
  const int w0 = (wid & 3) << 4;
  const int h  = (wid >> 2) & 63;
  const int n  = wid >> 8;
  const int col = lane & 15, hi = lane >> 4;
  const __bf16* sb = (const __bf16*)sB4;

  const v8f zf = {0.f,0.f,0.f,0.f,0.f,0.f,0.f,0.f};
  v8f acc[4] = {zf, zf, zf, zf};

  for (int t = 0; t < 9; ++t) {
#if HAVE_TDM
    if (threadIdx.x == 0)
      tdm_stage_16k((const void*)(fw + (size_t)t*8192), (void*)sB4);
#else
    coop_stage_16k((const void*)(fw + (size_t)t*8192), sB4);
#endif
    __syncthreads();

    const int dy = t/3 - 1, dx = t%3 - 1;
    const int hh = h + dy;
    if ((unsigned)hh < (unsigned)HH) {        // wave-uniform
      const int ww = w0 + col + dx;           // A-row pixel for this lane
      const bool ok = (unsigned)ww < (unsigned)WW;
      const float* rx = combx + ((size_t)((n*HH + hh)*WW) + ww)*CC;
      const float* ry = comby + ((size_t)((n*HH + hh)*WW) + ww)*CC;
      v16bf afr[4];
#pragma unroll
      for (int kc = 0; kc < 4; ++kc) {        // K = 128 = 4 x 32
        BF16x16 u;
        if (ok) {
          const float* src = (kc < 2) ? rx  : ry;
          const float* sS  = (kc < 2) ? nxs : nys;
          const float* sB  = (kc < 2) ? nxb : nyb;
          const int base = (kc & 1)*32 + 8*hi;
#pragma unroll
          for (int j = 0; j < 8; ++j) {
            const int c0 = base + j, c1 = base + 16 + j;
            u.v[j]     = (__bf16)fmaxf(src[c0]*sS[c0] + sB[c0], 0.f);  // fused BN+ReLU
            u.v[j + 8] = (__bf16)fmaxf(src[c1]*sS[c1] + sB[c1], 0.f);
          }
        } else {
          u.q[0] = make_uint4(0,0,0,0);
          u.q[1] = make_uint4(0,0,0,0);
        }
        afr[kc] = u.v;
      }
#pragma unroll
      for (int nb = 0; nb < 4; ++nb)
#pragma unroll
        for (int kc = 0; kc < 4; ++kc)
          acc[nb] = wmma_bf16(afr[kc],
                              load_b_frag(sb + (kc*4 + nb)*512 + lane*16),
                              acc[nb]);
    }
    __syncthreads();
  }

  // final BN + ReLU, store NCHW
#pragma unroll
  for (int nb = 0; nb < 4; ++nb) {
    const int oc = nb*16 + col;
    const float sc = bns[oc], bi = bnb[oc];
#pragma unroll
    for (int v = 0; v < 8; ++v) {
      const int m = v + 8*hi;
      out[(((size_t)n*CC + oc)*HH + h)*WW + (w0 + m)] = fmaxf(acc[nb][v]*sc + bi, 0.f);
    }
  }
}

// ---------------- launcher ----------------
extern "C" void kernel_launch(void* const* d_in, const int* in_sizes, int n_in,
                              void* d_out, int out_size, void* d_ws, size_t ws_size,
                              hipStream_t stream) {
  (void)in_sizes; (void)n_in; (void)out_size; (void)ws_size;
  const float* Y = (const float*)d_in[0];
  const float* X = (const float*)d_in[1];
  char* ws = (char*)d_ws;
  __bf16* act   = (__bf16*)(ws + ACT_OFF);
  __bf16* bw    = (__bf16*)(ws + BW_OFF);
  float*  combx = (float*) (ws + COMBX_OFF);
  float*  comby = (float*) (ws + COMBY_OFF);
  float*  xtp   = (float*) (ws + XTP_OFF);
  float*  ytp   = (float*) (ws + YTP_OFF);
  __bf16* fwb   = (__bf16*)(ws + FW_OFF);

  static const int hKS[5] = {5,5,3,3,3};
  static const int hRS[5] = {1,2,3,4,5};
  static const int hSBASE[5] = {0, 204800, 409600, 483328, 557056};

  dim3 b256(256);

  // zero comb accumulators + padded borders of x/y copies (one contiguous region)
  (void)hipMemsetAsync(ws + COMBX_OFF, 0, 9437184, stream);

  k_transpose_pad<<<ELEMS/256, b256, 0, stream>>>(X, xtp);
  k_transpose_pad<<<ELEMS/256, b256, 0, stream>>>(Y, ytp);

  for (int s = 0; s < 5; ++s) {
    for (int sd = 0; sd < 2; ++sd) {        // sd 0='y', 1='x' (params dict order)
      const int g = s*2 + sd;
      const float* dwp = (const float*)d_in[2 + g*5 + 0];
      const float* bns = (const float*)d_in[2 + g*5 + 1];
      const float* bnb = (const float*)d_in[2 + g*5 + 2];
      const float* pww = (const float*)d_in[2 + g*5 + 3];
      const float* inT = (sd == 0) ? ytp : xtp;
      k_prep_act<<<ELEMS/256, b256, 0, stream>>>(
          inT, dwp, bns, bnb, act + (size_t)g*ELEMS, hKS[s], hRS[s]);
      const int nel = hKS[s]*hKS[s]*4096;
      k_prep_bw<<<nel/256, b256, 0, stream>>>(pww, bw + hSBASE[s] + sd*4096, hKS[s]);
    }
  }
  k_prep_fw<<<288, b256, 0, stream>>>((const float*)d_in[56], fwb);

  DfnArgs da;
  da.act = act; da.bw = bw; da.xt = xtp; da.yt = ytp;
  da.combx = combx; da.comby = comby;
  for (int g = 0; g < 10; ++g) da.pwb[g] = (const float*)d_in[2 + g*5 + 4];
  k_dfn_fused<<<320, b256, 0, stream>>>(da);   // 5 scales x 64 tile-blocks, 8 waves each

  k_fuse_conv<<<64, b256, 0, stream>>>(
      combx, comby, fwb,
      (const float*)d_in[52], (const float*)d_in[53],
      (const float*)d_in[54], (const float*)d_in[55],
      (const float*)d_in[57], (const float*)d_in[58],
      (float*)d_out);
}
